// MPWCNet2_91216515432548
// MI455X (gfx1250) — matrix operand order repaired and checked by reading, not compiled, not run on hardware
//
#include <hip/hip_runtime.h>
#include <hip/hip_bf16.h>

typedef __attribute__((ext_vector_type(16))) _Float16     v16h;
typedef __attribute__((ext_vector_type(8)))  float        v8f;
typedef __attribute__((ext_vector_type(4)))  unsigned int u32x4;

#define DIV_FLOW_C 0.1f
#define MD 5

// ---- gfx1250 async global->LDS path (guarded; sync fallback keeps build green)
#define TRY_ASYNC 1
#if TRY_ASYNC && defined(__has_builtin)
#if __has_builtin(__builtin_amdgcn_global_load_async_to_lds_b128)
#define USE_ASYNC_LDS 1
#endif
#endif

// builtin parameter type (from hipcc diagnostic): int4 in AS1 / AS3
typedef int i32x4 __attribute__((vector_size(16)));
typedef __attribute__((address_space(1))) i32x4 gi32x4;
typedef __attribute__((address_space(3))) i32x4 li32x4;

__device__ __forceinline__ void async_wait_all() {
#if defined(USE_ASYNC_LDS)
#if __has_builtin(__builtin_amdgcn_s_wait_asynccnt)
    __builtin_amdgcn_s_wait_asynccnt(0);
#else
    asm volatile("s_wait_asynccnt 0x0" ::: "memory");
#endif
#endif
}

// ---------------------------------------------------------------------------
// Weight pre-pack: f32 [OC, Ktot] -> f16 fragment-ordered tiles.
// Layout: [fp][kc][which(2)][lane(32)][j(16)] halves; one (fp,kc) block = 2KB,
// exactly the A-operand VGPR layout for two 16-oc WMMA fragments.
// ---------------------------------------------------------------------------
__global__ void k_packw(const float* __restrict__ Wt, _Float16* __restrict__ Pk,
                        int OC, int Ktot, int nK32, int nFp) {
    int i = blockIdx.x * blockDim.x + threadIdx.x;
    int total = nFp * nK32 * 1024;
    if (i >= total) return;
    int fp = i / (nK32 * 1024);
    int r  = i - fp * (nK32 * 1024);
    int kc = r >> 10;
    int r2 = r & 1023;
    int which = r2 >> 9;
    int l = (r2 >> 4) & 31;
    int j = r2 & 15;
    int oc = fp * 32 + which * 16 + (l & 15);
    int k  = ((l & 16) ? 8 : 0) + (j & 7) + ((j >> 3) << 4);
    int kk = kc * 32 + k;
    float v = (oc < OC && kk < Ktot) ? Wt[(size_t)oc * Ktot + kk] : 0.f;
    Pk[i] = (_Float16)v;
}

// ---------------------------------------------------------------------------
// Implicit-GEMM conv2d on WMMA. Block = 128 thr = 4 waves.
// Block tile: 32 output channels x 64 pixels; each wave: 32 oc x 16 px
// -> 2 x v_wmma_f32_16x16x32_f16 per K-step.
// A (weights, f16 pre-packed) staged via double-buffered LDS, async when
// available. B (im2col) built directly in VGPRs: lane l of wave w owns the
// 16 K-halves of its own pixel column per the ISA B layout.
// ---------------------------------------------------------------------------
template <int KH, int KW>
__global__ __launch_bounds__(128)
void conv_wmma(const float* __restrict__ X, const _Float16* __restrict__ Pk,
               const float* __restrict__ Bias, float* __restrict__ Y,
               int Bn, int H, int W,
               int IC, int ICtot, int icOff,
               int OC, int OCtot, int ocOff,
               int nK32, int dil, int doRelu)
{
    __shared__ __align__(16) _Float16 Abuf[2][1024];

    const int tid  = threadIdx.x;
    const int wave = tid >> 5;
    const int lane = tid & 31;

    const int pxBase = blockIdx.x * 64;
    const int fp     = blockIdx.y;          // fragment-pair (32 oc)
    const int ocTile = fp * 32;

    constexpr int KHW = KH * KW;
    const int Ktot = IC * KHW;
    const int ph   = (KH - 1) * dil / 2;
    const int pw   = (KW - 1) * dil / 2;
    const int HW   = H * W;
    const int NPX  = Bn * HW;

    // hoisted per-thread pixel decomposition (this thread's B column)
    const int p   = pxBase + wave * 16 + (lane & 15);
    const bool pOK = (p < NPX);
    int b = 0, oy = 0, ox = 0;
    size_t baseX = 0;
    if (pOK) {
        b  = p / HW;
        int r0 = p - b * HW;
        oy = r0 / W; ox = r0 - oy * W;
        baseX = ((size_t)b * ICtot + icOff) * HW;
    }
    const int kB = (lane & 16) >> 1;        // this lane's K base within a chunk

    const _Float16* wbase = Pk + (size_t)fp * nK32 * 1024;

    auto stage = [&](int kc, int buf) {
        const _Float16* src = wbase + (size_t)kc * 1024 + tid * 8;  // 16B/thread
#if defined(USE_ASYNC_LDS)
        __builtin_amdgcn_global_load_async_to_lds_b128(
            (gi32x4*)(unsigned long long)(const void*)src,
            (li32x4*)(unsigned long long)(const void*)&Abuf[buf][tid * 8],
            0, 0);
#else
        __builtin_prefetch(src + 1024, 0, 1);
        *(u32x4*)&Abuf[buf][tid * 8] = *(const u32x4*)src;
#endif
    };

    v8f acc0 = {}, acc1 = {};

    stage(0, 0);
    int cur = 0;
    for (int kc = 0; kc < nK32; ++kc) {
        async_wait_all();
        __syncthreads();                     // Abuf[cur] ready for all waves
        if (kc + 1 < nK32) stage(kc + 1, cur ^ 1);  // overlap next copy

        // B fragment: built in registers (template-constant div/mod)
        v16h bv;
        const int kc32 = kc * 32 + kB;
#pragma unroll
        for (int j = 0; j < 16; ++j) {
            int kk = kc32 + (j & 7) + ((j >> 3) << 4);
            float v = 0.f;
            if (pOK && kk < Ktot) {
                int ic = kk / KHW;
                int rr = kk - ic * KHW;
                int dy = rr / KW, dx = rr - dy * KW;
                int iy = oy + dy * dil - ph;
                int ix = ox + dx * dil - pw;
                if (iy >= 0 && iy < H && ix >= 0 && ix < W)
                    v = X[baseX + (size_t)ic * HW + iy * W + ix];
            }
            bv[j] = (_Float16)v;
        }

        v16h av0 = *(const v16h*)&Abuf[cur][lane * 16];
        v16h av1 = *(const v16h*)&Abuf[cur][512 + lane * 16];
        acc0 = __builtin_amdgcn_wmma_f32_16x16x32_f16(false, av0, false, bv,
                                                      (short)0, acc0, false, false);
        acc1 = __builtin_amdgcn_wmma_f32_16x16x32_f16(false, av1, false, bv,
                                                      (short)0, acc1, false, false);
        __syncthreads();                     // reads done before buf reuse
        cur ^= 1;
    }

    // Epilogue. C/D: VGPR r -> M = r + (lane>=16 ? 8 : 0), N = lane & 15.
    if (pOK) {
        const int mh = (lane & 16) ? 8 : 0;
        const size_t outPix = (size_t)oy * W + ox;
#pragma unroll
        for (int r = 0; r < 8; ++r) {
            int oc0 = ocTile + mh + r;
            if (oc0 < OC) {
                float v = acc0[r] + Bias[oc0];
                if (doRelu) v = (v >= 0.f) ? v : 0.1f * v;
                Y[((size_t)b * OCtot + ocOff + oc0) * HW + outPix] = v;
            }
            int oc1 = ocTile + 16 + mh + r;
            if (oc1 < OC) {
                float v = acc1[r] + Bias[oc1];
                if (doRelu) v = (v >= 0.f) ? v : 0.1f * v;
                Y[((size_t)b * OCtot + ocOff + oc1) * HW + outPix] = v;
            }
        }
    }
}

// ---------------------------------------------------------------------------
// Elementwise / data-movement kernels
// ---------------------------------------------------------------------------
__global__ void k_zero(float* p, int n) {
    int i = blockIdx.x * blockDim.x + threadIdx.x;
    if (i < n) p[i] = 0.f;
}

__global__ void k_add(float* a, const float* __restrict__ b, int n) {
    int i = blockIdx.x * blockDim.x + threadIdx.x;
    if (i < n) a[i] += b[i];
}

__global__ void k_scale2(float* f, int B, int HW, float us, float vs) {
    int i = blockIdx.x * blockDim.x + threadIdx.x;
    int n = B * 2 * HW;
    if (i < n) {
        int c = (i / HW) & 1;
        f[i] *= c ? vs : us;
    }
}

__global__ void k_copych(const float* __restrict__ src, int C,
                         float* __restrict__ dst, int OCtot, int ocOff,
                         int B, int HW) {
    int i = blockIdx.x * blockDim.x + threadIdx.x;
    int n = B * C * HW;
    if (i < n) {
        int b = i / (C * HW);
        int r = i - b * C * HW;
        int c = r / HW;
        int p = r - c * HW;
        dst[((size_t)b * OCtot + ocOff + c) * HW + p] = src[i];
    }
}

__global__ void k_avgpool2(const float* __restrict__ x, float* __restrict__ y,
                           int B, int C, int H, int W) {
    int H2 = H >> 1, W2 = W >> 1;
    int i = blockIdx.x * blockDim.x + threadIdx.x;
    int n = B * C * H2 * W2;
    if (i < n) {
        int bc = i / (H2 * W2);
        int r  = i - bc * (H2 * W2);
        int oy = r / W2, ox = r - oy * W2;
        const float* s = x + (size_t)bc * H * W + (oy * 2) * W + ox * 2;
        y[i] = 0.25f * (s[0] + s[1] + s[W] + s[W + 1]);
    }
}

__global__ void k_costvol(const float* __restrict__ f1, const float* __restrict__ f2,
                          float* __restrict__ dst,
                          int B, int C, int H, int W, int OCtot, int ocOff) {
    int i = blockIdx.x * blockDim.x + threadIdx.x;
    int HW = H * W;
    int n = B * 121 * HW;
    if (i >= n) return;
    int b  = i / (121 * HW);
    int r  = i - b * (121 * HW);
    int ij = r / HW;
    int p  = r - ij * HW;
    int y  = p / W, x = p - y * W;
    int dy = ij / 11 - MD;
    int dx = ij % 11 - MD;
    int yy = y + dy, xx = x + dx;
    float s = 0.f;
    if (yy >= 0 && yy < H && xx >= 0 && xx < W) {
        const float* a = f1 + ((size_t)b * C) * HW + y * W + x;
        const float* c = f2 + ((size_t)b * C) * HW + yy * W + xx;
        for (int k = 0; k < C; ++k) s += a[(size_t)k * HW] * c[(size_t)k * HW];
        s /= (float)C;
    }
    s = (s >= 0.f) ? s : 0.1f * s;
    dst[((size_t)b * OCtot + ocOff + ij) * HW + y * W + x] = s;
}

__global__ void k_resize(const float* __restrict__ src, float* __restrict__ dst,
                         int B, int C, int Hs, int Ws, int Ht, int Wt, float outScale) {
    int i = blockIdx.x * blockDim.x + threadIdx.x;
    int n = B * C * Ht * Wt;
    if (i >= n) return;
    int bc = i / (Ht * Wt);
    int r  = i - bc * (Ht * Wt);
    int y  = r / Wt, x = r - y * Wt;
    float sy = (Ht == 1) ? 0.f : (float)(Hs - 1) / (float)(Ht - 1);
    float sx = (Wt == 1) ? 0.f : (float)(Ws - 1) / (float)(Wt - 1);
    float cy = y * sy, cx = x * sx;
    int y0 = (int)floorf(cy); y0 = y0 < 0 ? 0 : (y0 > Hs - 1 ? Hs - 1 : y0);
    int x0 = (int)floorf(cx); x0 = x0 < 0 ? 0 : (x0 > Ws - 1 ? Ws - 1 : x0);
    int y1 = y0 + 1 > Hs - 1 ? Hs - 1 : y0 + 1;
    int x1 = x0 + 1 > Ws - 1 ? Ws - 1 : x0 + 1;
    float wy = cy - (float)y0, wx = cx - (float)x0;
    const float* s = src + (size_t)bc * Hs * Ws;
    float v = s[y0 * Ws + x0] * (1.f - wy) * (1.f - wx)
            + s[y0 * Ws + x1] * (1.f - wy) * wx
            + s[y1 * Ws + x0] * wy * (1.f - wx)
            + s[y1 * Ws + x1] * wy * wx;
    dst[i] = v * outScale;
}

__global__ void k_warp(const float* __restrict__ x, const float* __restrict__ flow,
                       float* __restrict__ out,
                       int B, int C, int H, int W, int Him, int Wim, float div_flow) {
    int i = blockIdx.x * blockDim.x + threadIdx.x;
    int HW = H * W;
    int n = B * C * HW;
    if (i >= n) return;
    int b  = i / (C * HW);
    int r  = i - b * (C * HW);
    int c  = r / HW;
    int p  = r - c * HW;
    int y  = p / W, xp = p - y * W;

    float fw = flow[((size_t)b * 2 + 0) * HW + p] * 2.f / ((float)(Wim - 1) * div_flow);
    float fh = flow[((size_t)b * 2 + 1) * HW + p] * 2.f / ((float)(Him - 1) * div_flow);
    float gx = -1.f + 2.f * (float)xp / (float)(W - 1) + fw;
    float gy = -1.f + 2.f * (float)y  / (float)(H - 1) + fh;
    float px = (gx + 1.f) * 0.5f * (float)(W - 1);
    float py = (gy + 1.f) * 0.5f * (float)(H - 1);
    float x0f = floorf(px), y0f = floorf(py);
    int   x0  = (int)x0f, y0 = (int)y0f;
    float wx1 = px - x0f, wx0 = 1.f - wx1;
    float wy1 = py - y0f, wy0 = 1.f - wy1;

    const float* s = x + ((size_t)b * C + c) * HW;
    float v = 0.f, m = 0.f;
#pragma unroll
    for (int t = 0; t < 4; ++t) {
        int yy = y0 + (t >> 1);
        int xx = x0 + (t & 1);
        float wgt = ((t >> 1) ? wy1 : wy0) * ((t & 1) ? wx1 : wx0);
        bool ok = (xx >= 0 && xx <= W - 1 && yy >= 0 && yy <= H - 1);
        if (ok) { v += s[yy * W + xx] * wgt; m += wgt; }
    }
    out[i] = (m >= 0.9999f) ? v : 0.f;
}

// ---------------------------------------------------------------------------
// Host orchestration
// ---------------------------------------------------------------------------
extern "C" void kernel_launch(void* const* d_in, const int* in_sizes, int n_in,
                              void* d_out, int out_size, void* d_ws, size_t ws_size,
                              hipStream_t stream) {
    const int B = 4, Him = 256, Wim = 256;

    const float* in1 = (const float*)d_in[0];
    const float* in2 = (const float*)d_in[1];
    int pi = 2;
    const float *fpw[6][2], *fpb[6][2];
    for (int l = 0; l < 6; ++l)
        for (int c = 0; c < 2; ++c) { fpw[l][c] = (const float*)d_in[pi++]; fpb[l][c] = (const float*)d_in[pi++]; }
    const float *c11w[5], *c11b[5];
    for (int l = 0; l < 5; ++l) { c11w[l] = (const float*)d_in[pi++]; c11b[l] = (const float*)d_in[pi++]; }
    const float *few[6], *feb[6];
    for (int l = 0; l < 6; ++l) { few[l] = (const float*)d_in[pi++]; feb[l] = (const float*)d_in[pi++]; }
    const float *cnw[7], *cnb[7];
    for (int l = 0; l < 7; ++l) { cnw[l] = (const float*)d_in[pi++]; cnb[l] = (const float*)d_in[pi++]; }
    (void)ws_size; (void)in_sizes; (void)n_in; (void)out_size;

    char*  ws  = (char*)d_ws;
    size_t off = 0;
    auto allocB = [&](size_t bytes) -> void* {
        void* p = (void*)(ws + off);
        off = (off + bytes + 255) & ~(size_t)255;
        return p;
    };
    auto allocF = [&](size_t n) -> float* { return (float*)allocB(n * sizeof(float)); };

    // ---- pre-pack all conv weights to f16 fragment tiles ----
    auto packW = [&](const float* w, int OC, int Ktot) -> const _Float16* {
        int nK32 = (Ktot + 31) / 32, nFp = (OC + 31) / 32;
        size_t total = (size_t)nFp * nK32 * 1024;
        _Float16* p = (_Float16*)allocB(total * sizeof(_Float16));
        k_packw<<<dim3((unsigned)((total + 255) / 256)), 256, 0, stream>>>(w, p, OC, Ktot, nK32, nFp);
        return p;
    };

    const int fC[6]  = {16, 32, 64, 96, 128, 196};
    const int fIC[6] = {3, 16, 32, 64, 96, 128};
    const int fH[6]  = {128, 128, 64, 32, 16, 8};
    const int Cl[5]  = {196, 128, 96, 64, 32};
    const int feIC[6] = {155, 283, 411, 507, 571, 603};
    const int feOC[6] = {128, 128, 96, 64, 32, 2};
    const int cnIC[7] = {605, 128, 128, 128, 96, 64, 32};
    const int cnOC[7] = {128, 128, 128, 96, 64, 32, 2};

    const _Float16 *fpP[6][2], *c11P[5], *feP[6], *cnP[7];
    for (int l = 0; l < 6; ++l) {
        fpP[l][0] = packW(fpw[l][0], fC[l], fIC[l] * ((l == 0) ? 15 : 9));
        fpP[l][1] = packW(fpw[l][1], fC[l], fC[l] * 9);
    }
    for (int l = 0; l < 5; ++l) c11P[l] = packW(c11w[l], 32, Cl[l]);
    for (int l = 0; l < 6; ++l) feP[l]  = packW(few[l], feOC[l], feIC[l] * 9);
    for (int l = 0; l < 7; ++l) cnP[l]  = packW(cnw[l], cnOC[l], cnIC[l] * 9);

    // ---- activation buffers ----
    float* tmpA = allocF((size_t)B * 16 * 256 * 256);
    float* tmpB = allocF((size_t)B * 16 * 128 * 128);
    float *feat1[6], *feat2[6];
    for (int l = 0; l < 6; ++l) feat1[l] = allocF((size_t)B * fC[l] * fH[l] * fH[l]);
    for (int l = 0; l < 6; ++l) feat2[l] = allocF((size_t)B * fC[l] * fH[l] * fH[l]);
    float* big  = allocF((size_t)B * 605 * 128 * 128);
    float* cnA  = allocF((size_t)B * 128 * 128 * 128);
    float* cnB  = allocF((size_t)B * 128 * 128 * 128);
    float* flowA   = allocF((size_t)B * 2 * 128 * 128);
    float* flowB   = allocF((size_t)B * 2 * 128 * 128);
    float* flowRes = allocF((size_t)B * 2 * 128 * 128);
    float* x2w     = allocF((size_t)B * 32 * 128 * 128);

    auto conv = [&](const float* X, int ICtot, int icOff, int IC,
                    const _Float16* Pk, const float* Bias,
                    float* Y, int OCtot, int ocOff, int OC,
                    int H, int W, int KH, int KW, int dil, int relu) {
        int Ktot = IC * KH * KW;
        int nK32 = (Ktot + 31) / 32;
        dim3 grid((B * H * W + 63) / 64, (OC + 31) / 32);
        if (KH == 5)
            conv_wmma<5, 3><<<grid, 128, 0, stream>>>(X, Pk, Bias, Y, B, H, W,
                IC, ICtot, icOff, OC, OCtot, ocOff, nK32, dil, relu);
        else if (KH == 3)
            conv_wmma<3, 3><<<grid, 128, 0, stream>>>(X, Pk, Bias, Y, B, H, W,
                IC, ICtot, icOff, OC, OCtot, ocOff, nK32, dil, relu);
        else
            conv_wmma<1, 1><<<grid, 128, 0, stream>>>(X, Pk, Bias, Y, B, H, W,
                IC, ICtot, icOff, OC, OCtot, ocOff, nK32, dil, relu);
    };
    auto ew = [&](int n) { return dim3((n + 255) / 256); };

    // ---------------- feature pyramids ----------------
    auto buildPyr = [&](const float* img, float** feats) {
        const float* cur = img;
        int curH = 256, curC = 3;
        for (int l = 0; l < 6; ++l) {
            int oc = fC[l];
            int kh = (l == 0) ? 5 : 3;
            conv(cur, curC, 0, curC, fpP[l][0], fpb[l][0], tmpA, oc, 0, oc, curH, curH, kh, 3, 1, 1);
            const float* post = tmpA;
            int ph2 = curH;
            if (l != 1) {
                ph2 = curH / 2;
                k_avgpool2<<<ew(B * oc * ph2 * ph2), 256, 0, stream>>>(tmpA, tmpB, B, oc, curH, curH);
                post = tmpB;
            }
            conv(post, oc, 0, oc, fpP[l][1], fpb[l][1], feats[l], oc, 0, oc, ph2, ph2, 3, 3, 1, 1);
            cur = feats[l]; curH = ph2; curC = oc;
        }
    };
    buildPyr(in1, feat1);
    buildPyr(in2, feat2);

    const int Hl[5] = {8, 16, 32, 64, 128};
    const int OT = 605;
    // concat offsets: o5@0 o4@32 o3@96 o2@192 o1@320 corr@448 x1_1@569 flowL@601 flow@603

    float* flow = flowA;
    float* flowNxt = flowB;

    for (int l = 0; l < 5; ++l) {
        int Hc = Hl[l], Wc = Hc, HW = Hc * Wc;
        int C  = Cl[l];
        const float* x1 = feat1[5 - l];
        const float* x2 = feat2[5 - l];
        const float* x2warp = x2;

        if (l == 0) {
            k_zero<<<ew(B * 2 * HW), 256, 0, stream>>>(flow, B * 2 * HW);
        } else {
            int Hp = Hl[l - 1];
            k_resize<<<ew(B * 2 * HW), 256, 0, stream>>>(flow, flowNxt, B, 2, Hp, Hp, Hc, Wc, 1.f);
            float* t = flow; flow = flowNxt; flowNxt = t;
            k_warp<<<ew(B * C * HW), 256, 0, stream>>>(x2, flow, x2w, B, C, Hc, Wc, Him, Wim, DIV_FLOW_C);
            x2warp = x2w;
        }

        k_costvol<<<ew(B * 121 * HW), 256, 0, stream>>>(x1, x2warp, big, B, C, Hc, Wc, OT, 448);

        {
            float us = (float)Wc / ((float)Wim * DIV_FLOW_C);
            float vs = (float)Hc / ((float)Him * DIV_FLOW_C);
            k_scale2<<<ew(B * 2 * HW), 256, 0, stream>>>(flow, B, HW, us, vs);
        }
        k_copych<<<ew(B * 2 * HW), 256, 0, stream>>>(flow, 2, big, OT, 601, B, HW);

        conv(x1, C, 0, C, c11P[l], c11b[l], big, OT, 569, 32, Hc, Wc, 1, 1, 1, 1);

        conv(big, OT, 448, 155, feP[0], feb[0], big, OT, 320, 128, Hc, Wc, 3, 3, 1, 1);
        conv(big, OT, 320, 283, feP[1], feb[1], big, OT, 192, 128, Hc, Wc, 3, 3, 1, 1);
        conv(big, OT, 192, 411, feP[2], feb[2], big, OT,  96,  96, Hc, Wc, 3, 3, 1, 1);
        conv(big, OT,  96, 507, feP[3], feb[3], big, OT,  32,  64, Hc, Wc, 3, 3, 1, 1);
        conv(big, OT,  32, 571, feP[4], feb[4], big, OT,   0,  32, Hc, Wc, 3, 3, 1, 1);
        conv(big, OT,   0, 603, feP[5], feb[5], flowRes, 2, 0, 2, Hc, Wc, 3, 3, 1, 0);

        k_add<<<ew(B * 2 * HW), 256, 0, stream>>>(flow, flowRes, B * 2 * HW);
        k_copych<<<ew(B * 2 * HW), 256, 0, stream>>>(flow, 2, big, OT, 603, B, HW);

        conv(big, OT, 0, 605, cnP[0], cnb[0], cnA, 128, 0, 128, Hc, Wc, 3, 3,  1, 1);
        conv(cnA, 128, 0, 128, cnP[1], cnb[1], cnB, 128, 0, 128, Hc, Wc, 3, 3,  2, 1);
        conv(cnB, 128, 0, 128, cnP[2], cnb[2], cnA, 128, 0, 128, Hc, Wc, 3, 3,  4, 1);
        conv(cnA, 128, 0, 128, cnP[3], cnb[3], cnB,  96, 0,  96, Hc, Wc, 3, 3,  8, 1);
        conv(cnB,  96, 0,  96, cnP[4], cnb[4], cnA,  64, 0,  64, Hc, Wc, 3, 3, 16, 1);
        conv(cnA,  64, 0,  64, cnP[5], cnb[5], cnB,  32, 0,  32, Hc, Wc, 3, 3,  1, 1);
        conv(cnB,  32, 0,  32, cnP[6], cnb[6], flowRes, 2, 0, 2, Hc, Wc, 3, 3,  1, 0);

        k_add<<<ew(B * 2 * HW), 256, 0, stream>>>(flow, flowRes, B * 2 * HW);

        {
            float us = ((float)Wim * DIV_FLOW_C) / (float)Wc;
            float vs = ((float)Him * DIV_FLOW_C) / (float)Hc;
            k_scale2<<<ew(B * 2 * HW), 256, 0, stream>>>(flow, B, HW, us, vs);
        }
    }

    k_resize<<<ew(B * 2 * Him * Wim), 256, 0, stream>>>(
        flow, (float*)d_out, B, 2, 128, 128, Him, Wim, 1.f / DIV_FLOW_C);
}
